// SS2D_12902081757680
// MI455X (gfx1250) — matrix operand order repaired and checked by reading, not compiled, not run on hardware
//
#include <hip/hip_runtime.h>
#include <math.h>

// ---------------- problem constants ----------------
#define NB   8
#define NH   64
#define NW   64
#define NC   96
#define NL   4096      // NH*NW
#define DI   192       // d_inner
#define DST  16        // d_state
#define DTR  6         // dt_rank
#define NCP  48        // padded projection cols: dts[0..5], pad, B[8..23], C[24..39], pad

typedef __attribute__((ext_vector_type(16))) __bf16          v16bf;
typedef __attribute__((ext_vector_type(16))) unsigned short  v16u;
typedef __attribute__((ext_vector_type(8)))  unsigned short  v8u;
typedef __attribute__((ext_vector_type(8)))  float           v8f;

__device__ __forceinline__ unsigned short f2bf(float f) {
  union { float f; unsigned u; } v; v.f = f;
  unsigned r = v.u + 0x7FFFu + ((v.u >> 16) & 1u);   // round-to-nearest-even
  return (unsigned short)(r >> 16);
}
__device__ __forceinline__ float siluf(float x) { return x / (1.f + __expf(-x)); }
__device__ __forceinline__ float softplusf(float x) { return (x > 20.f) ? x : log1pf(__expf(x)); }

// ---------------- elementwise helpers ----------------
__global__ void k_cvt_bf16(const float* __restrict__ s, unsigned short* __restrict__ d, int n) {
  int i = blockIdx.x * blockDim.x + threadIdx.x;
  if (i < n) d[i] = f2bf(s[i]);
}
__global__ void k_zero_f32(float* __restrict__ p, int n) {
  int i = blockIdx.x * blockDim.x + threadIdx.x;
  if (i < n) p[i] = 0.f;
}
// pack weights (N,K) fp32 -> bf16 WMMA fragment order [(n/16)][n%16][k]
__global__ void k_pack_w(const float* __restrict__ src, unsigned short* __restrict__ dst,
                         int N, int K) {
  int i = blockIdx.x * blockDim.x + threadIdx.x;
  if (i >= N * K) return;
  int k = i % K, n = i / K;
  dst[((size_t)(n >> 4) * 16 + (n & 15)) * K + k] = f2bf(src[(size_t)n * K + k]);
}
// pack x_proj_w[k2 in {0,1}] (38,192) into fragment order with remapped padded cols:
// c' 0..5 = dts rows, 6..7 = 0, 8..23 = B rows, 24..39 = C rows, 40..47 = 0
__global__ void k_pack_xpw(const float* __restrict__ xpw, unsigned short* __restrict__ dst) {
  int i = blockIdx.x * blockDim.x + threadIdx.x;        // 2*48*192
  if (i >= 2 * NCP * DI) return;
  int k2 = i / (NCP * DI);
  int rem = i % (NCP * DI);
  int cp = rem / DI, d = rem % DI;
  bool valid = (cp < DTR) || (cp >= 8 && cp < 8 + 2 * DST);
  int c = (cp < DTR) ? cp : cp - 2;
  float v = valid ? xpw[((size_t)k2 * (DTR + 2 * DST) + c) * DI + d] : 0.f;
  dst[(size_t)k2 * NCP * DI + ((size_t)(cp >> 4) * 16 + (cp & 15)) * DI + d] = f2bf(v);
}

__device__ __forceinline__ v16u load_afrag(const unsigned short* Arow, int ka) {
  v8u alo = *(const v8u*)(Arow + ka);
  v8u ahi = *(const v8u*)(Arow + ka + 16);
  return __builtin_shufflevector(alo, ahi, 0, 1, 2, 3, 4, 5, 6, 7,
                                 8, 9, 10, 11, 12, 13, 14, 15);
}

// ---------------- bf16 WMMA GEMM: C(M,N) = A(M,K) * W(N,K)^T ----------------
// one wave per 16x(16*NT) output tile; A fragment reused across NT WMMAs
template <int NT>
__global__ void k_gemm_bf16(const unsigned short* __restrict__ A,
                            const unsigned short* __restrict__ Bp,  // packed fragments
                            float* __restrict__ Co, int M, int N, int K) {
  int gtid = blockIdx.x * blockDim.x + threadIdx.x;
  int wave = gtid >> 5, lane = gtid & 31;
  int groupsN = (N >> 4) / NT;
  int tilesM = M >> 4;
  if (wave >= tilesM * groupsN) return;
  int tm = wave / groupsN, tn0 = (wave % groupsN) * NT;
  int half = lane >> 4, l16 = lane & 15;
  const unsigned short* Arow = A + (size_t)(tm * 16 + l16) * K;
  v8f acc[NT];
#pragma unroll
  for (int t = 0; t < NT; ++t) acc[t] = (v8f){};
  for (int k0 = 0; k0 < K; k0 += 32) {
    v16u a = load_afrag(Arow, k0 + half * 8);
    int kb = k0 + lane;
#pragma unroll
    for (int t = 0; t < NT; ++t) {
      v16u b = load_afrag(Bp + ((size_t)(tn0 + t) * 16 + l16) * K, k0 + half * 8);
      acc[t] = __builtin_amdgcn_wmma_f32_16x16x32_bf16(false, __builtin_bit_cast(v16bf, a),
                                                       false, __builtin_bit_cast(v16bf, b),
                                                       (short)0, acc[t], false, false);
    }
  }
#pragma unroll
  for (int t = 0; t < NT; ++t)
#pragma unroll
    for (int v = 0; v < 8; ++v)
      Co[(size_t)(tm * 16 + half * 8 + v) * N + (tn0 + t) * 16 + l16] = acc[t][v];
}

// ---------------- x_dbl GEMM with gathered A rows (direction-1 transpose) ----------------
// rows r = k2*(B*L) + b*L + l ; A row = xcb[b, (k2? transpose(l) : l), :] ; N=48 (3 tiles)
__global__ void k_gemm_xdbl(const unsigned short* __restrict__ xcb,
                            const unsigned short* __restrict__ xpwp,
                            float* __restrict__ xdbl) {
  const int K = DI;
  int gtid = blockIdx.x * blockDim.x + threadIdx.x;
  int wave = gtid >> 5, lane = gtid & 31;
  int tilesM = (2 * NB * NL) >> 4;
  if (wave >= tilesM) return;
  int half = lane >> 4, l16 = lane & 15;
  int r = wave * 16 + l16;
  int k2 = r / (NB * NL);
  int rem = r - k2 * NB * NL;
  int b = rem / NL, l = rem - b * NL;
  int srcl = k2 ? (((l & 63) << 6) | (l >> 6)) : l;     // column-major scan order
  const unsigned short* Arow  = xcb  + (size_t)(b * NL + srcl) * DI;
  const unsigned short* Bbase = xpwp + (size_t)k2 * NCP * DI;
  v8f acc[3];
#pragma unroll
  for (int t = 0; t < 3; ++t) acc[t] = (v8f){};
  for (int k0 = 0; k0 < K; k0 += 32) {
    v16u a = load_afrag(Arow, k0 + half * 8);
    int kb = k0 + lane;
#pragma unroll
    for (int t = 0; t < 3; ++t) {
      v16u bb = load_afrag(Bbase + ((size_t)t * 16 + l16) * K, k0 + half * 8);
      acc[t] = __builtin_amdgcn_wmma_f32_16x16x32_bf16(false, __builtin_bit_cast(v16bf, a),
                                                       false, __builtin_bit_cast(v16bf, bb),
                                                       (short)0, acc[t], false, false);
    }
  }
#pragma unroll
  for (int t = 0; t < 3; ++t)
#pragma unroll
    for (int v = 0; v < 8; ++v)
      xdbl[(size_t)(wave * 16 + half * 8 + v) * NCP + t * 16 + l16] = acc[t][v];
}

// ---------------- depthwise 3x3 conv + bias + SiLU (writes f32 + bf16) ----------------
__global__ void k_dwconv_silu(const float* __restrict__ xz,
                              const float* __restrict__ cw,
                              const float* __restrict__ cb,
                              float* __restrict__ xc,
                              unsigned short* __restrict__ xcb) {
  int i = blockIdx.x * blockDim.x + threadIdx.x;       // NB*NL*DI
  if (i >= NB * NL * DI) return;
  int d = i % DI, l = (i / DI) % NL, b = i / (DI * NL);
  int h = l >> 6, w = l & 63;
  float acc = cb[d];
#pragma unroll
  for (int dh = -1; dh <= 1; ++dh) {
    int hh = h + dh;
    if ((unsigned)hh >= NH) continue;
#pragma unroll
    for (int dw = -1; dw <= 1; ++dw) {
      int ww = w + dw;
      if ((unsigned)ww >= NW) continue;
      float v = xz[(size_t)(b * NL + hh * NW + ww) * (2 * DI) + d];
      acc += v * cw[d * 9 + (dh + 1) * 3 + (dw + 1)];
    }
  }
  float s = siluf(acc);
  size_t o = (size_t)(b * NL + l) * DI + d;
  xc[o] = s;
  xcb[o] = f2bf(s);
}

// ---------------- selective scan: one thread per (b,k,d) channel ----------------
// Fuses dt projection + softplus, Ds*u skip, and 4-direction recombination (atomics).
__global__ void k_scan(const float* __restrict__ xc,
                       const float* __restrict__ xdbl,
                       const float* __restrict__ dtw,   // (4,DI,DTR)
                       const float* __restrict__ dtb,   // (4,DI)
                       const float* __restrict__ Alog,  // (4,DI,DST)
                       const float* __restrict__ Dsk,   // (4,DI)
                       float* __restrict__ yacc) {
  int d = threadIdx.x;                 // 0..191
  int bk = blockIdx.x;                 // 0..31
  int b = bk >> 2, k = bk & 3, k2 = k & 1;
  float Ar[DST], wdt[DTR];
#pragma unroll
  for (int n = 0; n < DST; ++n) Ar[n] = -__expf(Alog[((size_t)k * DI + d) * DST + n]);
#pragma unroll
  for (int r = 0; r < DTR; ++r) wdt[r] = dtw[((size_t)k * DI + d) * DTR + r];
  float bdt = dtb[k * DI + d];
  float dsd = Dsk[k * DI + d];
  float h[DST];
#pragma unroll
  for (int n = 0; n < DST; ++n) h[n] = 0.f;
  for (int l = 0; l < NL; ++l) {
    int lsrc = (k >= 2) ? (NL - 1 - l) : l;                       // flip for k=2,3
    int lu = k2 ? (((lsrc & 63) << 6) | (lsrc >> 6)) : lsrc;      // spatial index
    const float* crow = xdbl + (size_t)((k2 * NB + b) * NL + lsrc) * NCP;
    // vectorized broadcast loads of the projection row
    float4 q0 = *(const float4*)(crow);          // dts 0..3
    float2 q1 = *(const float2*)(crow + 4);      // dts 4..5
    float Bc[DST], Cc[DST];
    *(float4*)(Bc + 0)  = *(const float4*)(crow + 8);
    *(float4*)(Bc + 4)  = *(const float4*)(crow + 12);
    *(float4*)(Bc + 8)  = *(const float4*)(crow + 16);
    *(float4*)(Bc + 12) = *(const float4*)(crow + 20);
    *(float4*)(Cc + 0)  = *(const float4*)(crow + 24);
    *(float4*)(Cc + 4)  = *(const float4*)(crow + 28);
    *(float4*)(Cc + 8)  = *(const float4*)(crow + 32);
    *(float4*)(Cc + 12) = *(const float4*)(crow + 36);
    float u = xc[(size_t)(b * NL + lu) * DI + d];
    float xv = bdt + q0.x * wdt[0] + q0.y * wdt[1] + q0.z * wdt[2] +
               q0.w * wdt[3] + q1.x * wdt[4] + q1.y * wdt[5];
    float dt = softplusf(xv);
    float dtu = dt * u;
    float y = dsd * u;
#pragma unroll
    for (int n = 0; n < DST; ++n) {
      float dA = __expf(dt * Ar[n]);
      h[n] = dA * h[n] + dtu * Bc[n];
      y += h[n] * Cc[n];
    }
    atomicAdd(&yacc[(size_t)(b * NL + lu) * DI + d], y);
  }
}

// ---------------- LayerNorm + SiLU(z) gating -> bf16 (wave32 shuffle reduce) -------------
__global__ void k_ln_gate(const float* __restrict__ yacc,
                          const float* __restrict__ xz,
                          const float* __restrict__ g,
                          const float* __restrict__ be,
                          unsigned short* __restrict__ yb) {
  __shared__ float r1[DI / 32], r2[DI / 32];
  int bl = blockIdx.x;                 // 0..NB*NL-1
  int d = threadIdx.x;                 // 0..191
  float v = yacc[(size_t)bl * DI + d];
  float s1 = v, s2 = v * v;
#pragma unroll
  for (int off = 16; off >= 1; off >>= 1) {
    s1 += __shfl_xor(s1, off, 32);
    s2 += __shfl_xor(s2, off, 32);
  }
  if ((d & 31) == 0) { r1[d >> 5] = s1; r2[d >> 5] = s2; }
  __syncthreads();
  float sum = 0.f, sq = 0.f;
#pragma unroll
  for (int i = 0; i < DI / 32; ++i) { sum += r1[i]; sq += r2[i]; }
  float mu = sum * (1.f / DI);
  float var = sq * (1.f / DI) - mu * mu;
  float yn = (v - mu) * rsqrtf(var + 1e-5f) * g[d] + be[d];
  float z = xz[(size_t)bl * (2 * DI) + DI + d];
  yb[(size_t)bl * DI + d] = f2bf(yn * siluf(z));
}

// ---------------- launch ----------------
extern "C" void kernel_launch(void* const* d_in, const int* in_sizes, int n_in,
                              void* d_out, int out_size, void* d_ws, size_t ws_size,
                              hipStream_t stream) {
  (void)in_sizes; (void)n_in; (void)out_size; (void)ws_size;
  const float* x    = (const float*)d_in[0];
  const float* ipw  = (const float*)d_in[1];
  const float* cw   = (const float*)d_in[2];
  const float* cb   = (const float*)d_in[3];
  const float* xpw  = (const float*)d_in[4];
  const float* dtw  = (const float*)d_in[5];
  const float* dtb  = (const float*)d_in[6];
  const float* Alog = (const float*)d_in[7];
  const float* Dsk  = (const float*)d_in[8];
  const float* lng  = (const float*)d_in[9];
  const float* lnb  = (const float*)d_in[10];
  const float* opw  = (const float*)d_in[11];
  float* out = (float*)d_out;

  char* ws = (char*)d_ws;
  size_t off = 0;
  auto alloc = [&](size_t bytes) -> char* {
    char* p = ws + off;
    off = (off + bytes + 255) & ~(size_t)255;
    return p;
  };
  const int M = NB * NL;                                           // 32768 tokens
  unsigned short* xb    = (unsigned short*)alloc((size_t)M * NC * 2);
  unsigned short* ipwp  = (unsigned short*)alloc((size_t)2 * DI * NC * 2);
  unsigned short* opwp  = (unsigned short*)alloc((size_t)NC * DI * 2);
  unsigned short* xpwp  = (unsigned short*)alloc((size_t)2 * NCP * DI * 2);
  float*          xzb   = (float*)alloc((size_t)M * 2 * DI * 4);
  float*          xcf   = (float*)alloc((size_t)M * DI * 4);
  unsigned short* xcb   = (unsigned short*)alloc((size_t)M * DI * 2);
  float*          xdbl  = (float*)alloc((size_t)2 * M * NCP * 4);
  float*          yacc  = (float*)alloc((size_t)M * DI * 4);
  unsigned short* yb    = (unsigned short*)alloc((size_t)M * DI * 2);

  const int T = 256;
  // 1) conversions + weight fragment packing
  int nX = M * NC;
  k_cvt_bf16<<<(nX + T - 1) / T, T, 0, stream>>>(x, xb, nX);
  int nIp = 2 * DI * NC;
  k_pack_w<<<(nIp + T - 1) / T, T, 0, stream>>>(ipw, ipwp, 2 * DI, NC);
  int nOp = NC * DI;
  k_pack_w<<<(nOp + T - 1) / T, T, 0, stream>>>(opw, opwp, NC, DI);
  int nXp = 2 * NCP * DI;
  k_pack_xpw<<<(nXp + T - 1) / T, T, 0, stream>>>(xpw, xpwp);

  // 2) in_proj: xz = x @ in_proj_w^T   (M=32768, N=384, K=96), NT=4
  {
    int waves = (M / 16) * ((2 * DI / 16) / 4);
    k_gemm_bf16<4><<<(waves * 32 + T - 1) / T, T, 0, stream>>>(xb, ipwp, xzb, M, 2 * DI, NC);
  }
  // 3) depthwise conv + SiLU
  {
    int n = M * DI;
    k_dwconv_silu<<<(n + T - 1) / T, T, 0, stream>>>(xzb, cw, cb, xcf, xcb);
  }
  // 4) x_dbl for directions 0,1 (2,3 are flips)  (M'=65536, N=48, K=192)
  {
    int waves = (2 * M) / 16;
    k_gemm_xdbl<<<(waves * 32 + T - 1) / T, T, 0, stream>>>(xcb, xpwp, xdbl);
  }
  // 5) selective scan (fused dt-proj/softplus/Ds/recombine)
  {
    int n = M * DI;
    k_zero_f32<<<(n + T - 1) / T, T, 0, stream>>>(yacc, n);
    k_scan<<<NB * 4, DI, 0, stream>>>(xcf, xdbl, dtw, dtb, Alog, Dsk, yacc);
  }
  // 6) LayerNorm + gate
  k_ln_gate<<<M, DI, 0, stream>>>(yacc, xzb, lng, lnb, yb);
  // 7) out_proj: out = y @ out_proj_w^T  (M=32768, N=96, K=192), NT=3
  {
    int waves = (M / 16) * ((NC / 16) / 3);
    k_gemm_bf16<3><<<(waves * 32 + T - 1) / T, T, 0, stream>>>(yb, opwp, out, M, NC, DI);
  }
}
